// TransformerBlock_15315853377769
// MI455X (gfx1250) — compile-verified, matrix operand order checked by reading, not run on hardware
//
#include <hip/hip_runtime.h>
#include <stdint.h>

// ---------------- problem constants ----------------
#define DIM   128
#define NH    16
#define HD    8
#define SEQ   2048
#define BATCH 4
#define MTOT  (BATCH*SEQ)   // 8192 rows
#define HID   344
#define HIDP  352           // padded to 16
#define KCH   256           // attention key chunk staged by TDM

typedef __attribute__((ext_vector_type(16))) __bf16 v16bf;
typedef __attribute__((ext_vector_type(8)))  __bf16 v8bf;
typedef __attribute__((ext_vector_type(8)))  float  v8f;
typedef __attribute__((ext_vector_type(4)))  unsigned int u32x4;
typedef __attribute__((ext_vector_type(8)))  int i32x8;
typedef __attribute__((ext_vector_type(4)))  int i32x4;

// ---------------- WMMA + fragment helpers ----------------
static __device__ inline v8f wmma_bf(v16bf a, v16bf b, v8f c) {
  // D = A(16x32 bf16) * B(32x16 bf16) + C(16x16 f32)
  return __builtin_amdgcn_wmma_f32_16x16x32_bf16(false, a, false, b,
                                                 (short)0, c, false, false);
}

static __device__ inline v8f vz8() {
  v8f v;
#pragma unroll
  for (int i = 0; i < 8; ++i) v[i] = 0.0f;
  return v;
}
static __device__ inline v16bf vz16() {
  v16bf v;
#pragma unroll
  for (int i = 0; i < 16; ++i) v[i] = (__bf16)0.0f;
  return v;
}

// A fragment (16x32), source fp32 row-major. lane holds row (lane&15);
// K pattern: half=lane>>4 -> elems 0..7 = K k0..k0+7, elems 8..15 = K k0+16..k0+23
static __device__ inline v16bf afrag_f32(const float* row, int kb, int half) {
  const int k0 = kb + half * 8;
  v16bf a;
#pragma unroll
  for (int j = 0; j < 8; ++j) {
    a[j]     = (__bf16)row[k0 + j];
    a[j + 8] = (__bf16)row[k0 + 16 + j];
  }
  return a;
}

// A fragment, source bf16 row-major (16B-aligned rows) -> two b128 loads
static __device__ inline v16bf afrag_bf(const __bf16* row, int kb, int half) {
  const int k0 = kb + half * 8;
  v8bf lo = *(const v8bf*)(row + k0);
  v8bf hi = *(const v8bf*)(row + k0 + 16);
  v16bf a;
#pragma unroll
  for (int j = 0; j < 8; ++j) { a[j] = lo[j]; a[j + 8] = hi[j]; }
  return a;
}

// B fragment (32x16) from fp32 weight W[K x N] (row-major, ldn = N).
// lane holds column `col`; rows beyond kmax are zero-padded.
static __device__ inline v16bf bfrag_w(const float* W, int ldn, int kb, int col,
                                       int kmax, int half) {
  const int k0 = kb + half * 8;
  v16bf b;
#pragma unroll
  for (int j = 0; j < 8; ++j) {
    const int ka = k0 + j, kc = k0 + 16 + j;
    b[j]     = (ka < kmax) ? (__bf16)W[(size_t)ka * ldn + col] : (__bf16)0.0f;
    b[j + 8] = (kc < kmax) ? (__bf16)W[(size_t)kc * ldn + col] : (__bf16)0.0f;
  }
  return b;
}

// ---------------- Tensor Data Mover: 1-D tile load (global -> LDS) ----------
// Descriptor packing per CDNA5 ISA §8.3/8.4 (D# group0 128b, group1 256b).
// 4-byte elements; nbytes must be a multiple of 4.
static __device__ inline void tdm_load_1d(uint32_t lds_off, const void* gptr,
                                          uint32_t nbytes) {
  const uint64_t ga = (uint64_t)(uintptr_t)gptr;
  const uint32_t elems = nbytes >> 2;          // data_size = 4B
  u32x4 g0;
  g0[0] = 1u;                                  // count=1 (valid user descriptor)
  g0[1] = lds_off;                             // lds_addr (bytes)
  g0[2] = (uint32_t)ga;                        // global_addr[31:0]
  g0[3] = (uint32_t)((ga >> 32) & 0x01FFFFFFu) // global_addr[56:32]
        | (2u << 30);                          // type = 2 ("image")
  i32x8 g1;
  g1[0] = (int)(2u << 16);                     // workgroup_mask=0, data_size=2 (4B)
  g1[1] = (int)((elems & 0xFFFFu) << 16);      // tensor_dim0[15:0] in bits 63:48
  g1[2] = (int)((elems >> 16) | (1u << 16));   // tensor_dim0[31:16] | tensor_dim1=1
  g1[3] = (int)((elems & 0xFFFFu) << 16);      // tile_dim0 = elems
  g1[4] = 1;                                   // tile_dim1=1, tile_dim2=0
  g1[5] = (int)elems;                          // tensor_dim0_stride[31:0]
  g1[6] = 0;                                   // stride hi / dim1 stride lo
  g1[7] = 0;
  i32x4 gz4;
  gz4[0] = 0; gz4[1] = 0; gz4[2] = 0; gz4[3] = 0;   // groups 2/3 unused (<=2D)
  i32x8 gz8;
#pragma unroll
  for (int i = 0; i < 8; ++i) gz8[i] = 0;
  __builtin_amdgcn_tensor_load_to_lds(g0, g1, gz4, gz4, gz8, 0);
}

// ---------------- 1) fused QKV projection ----------------
// x[8192,128] f32 @ {wq,wk,wv}[128,128] -> Q/K/V [B,H,S,8] bf16
__global__ void __launch_bounds__(256) k_qkv(const float* __restrict__ x,
                                             const float* __restrict__ wq,
                                             const float* __restrict__ wk,
                                             const float* __restrict__ wv,
                                             __bf16* __restrict__ Q,
                                             __bf16* __restrict__ K,
                                             __bf16* __restrict__ V) {
  const int wave = threadIdx.x >> 5, lane = threadIdx.x & 31;
  const int half = lane >> 4, l16 = lane & 15;
  const int tile = blockIdx.x * 8 + wave;
  if (tile >= (MTOT / 16) * 24) return;        // 512 row-tiles x (3 mats x 8 col-tiles)
  const int rt = tile & 511;
  const int t2 = tile >> 9;
  const int mat = t2 >> 3, ct = t2 & 7;
  const float* W = (mat == 0) ? wq : (mat == 1) ? wk : wv;
  __bf16* Out    = (mat == 0) ? Q  : (mat == 1) ? K  : V;

  const float* xrow = x + (size_t)(rt * 16 + l16) * DIM;
  const int col = ct * 16 + l16;
  v8f acc = vz8();
#pragma unroll
  for (int f = 0; f < 4; ++f) {
    v16bf a  = afrag_f32(xrow, f * 32, half);
    v16bf bw = bfrag_w(W, DIM, f * 32, col, DIM, half);
    acc = wmma_bf(a, bw, acc);
  }
  const int head = col >> 3, d = col & 7;
#pragma unroll
  for (int r = 0; r < 8; ++r) {
    const int grow = rt * 16 + half * 8 + r;
    const int b = grow >> 11, srow = grow & (SEQ - 1);
    Out[(((size_t)(b * NH + head)) * SEQ + srow) * HD + d] = (__bf16)acc[r];
  }
}

// ---------------- 2) flash attention with TDM-staged K/V ----------------
// grid (S/128, B*H); 8 waves/block, each wave owns a 16-row query tile.
// Wave 0 drives the Tensor Data Mover: double-buffered 256-key K and V
// chunks land in LDS asynchronously (TENSORcnt) while all waves compute.
__global__ void __launch_bounds__(256) k_attn(const __bf16* __restrict__ Q,
                                              const __bf16* __restrict__ K,
                                              const __bf16* __restrict__ V,
                                              __bf16* __restrict__ attn) {
  const int bh = blockIdx.y;
  const int wave = threadIdx.x >> 5, lane = threadIdx.x & 31;
  const int half = lane >> 4, l16 = lane & 15;
  const int qbase = blockIdx.x * 128 + wave * 16;
  const __bf16* Qh = Q + (size_t)bh * SEQ * HD;
  const __bf16* Kh = K + (size_t)bh * SEQ * HD;
  const __bf16* Vh = V + (size_t)bh * SEQ * HD;

  __shared__ __bf16 kbuf[2][KCH * HD];         // 2 x 4KB
  __shared__ __bf16 vbuf[2][KCH * HD];         // 2 x 4KB
  __shared__ __bf16 p_lds[8][16 * 32];         // per-wave prob tile (transpose buffer)

  // Q fragment: A(16x32), only d=0..7 real -> zeros elsewhere
  v16bf qf = vz16();
  if (half == 0) {
    v8bf qv = *(const v8bf*)(Qh + (size_t)(qbase + l16) * HD);
#pragma unroll
    for (int j = 0; j < 8; ++j) qf[j] = qv[j];
  }

  float mrow[8], lrow[8];
#pragma unroll
  for (int r = 0; r < 8; ++r) { mrow[r] = -1e30f; lrow[r] = 0.0f; }
  v8f oacc = vz8();

  // prologue: DMA chunk 0 into buffer 0 (K then V; TENSORcnt in-order per wave)
  if (wave == 0) {
    tdm_load_1d((uint32_t)(uintptr_t)&kbuf[0][0], Kh, KCH * HD * 2);
    tdm_load_1d((uint32_t)(uintptr_t)&vbuf[0][0], Vh, KCH * HD * 2);
  }

  for (int c = 0; c < SEQ / KCH; ++c) {
    if (wave == 0) __builtin_amdgcn_s_wait_tensorcnt(0);   // chunk c landed
    __syncthreads();                                       // visible to all waves
    if (wave == 0 && c + 1 < SEQ / KCH) {                  // prefetch chunk c+1
      const int nb = (c + 1) & 1;
      tdm_load_1d((uint32_t)(uintptr_t)&kbuf[nb][0],
                  Kh + (size_t)(c + 1) * KCH * HD, KCH * HD * 2);
      tdm_load_1d((uint32_t)(uintptr_t)&vbuf[nb][0],
                  Vh + (size_t)(c + 1) * KCH * HD, KCH * HD * 2);
    }
    const __bf16* kchb = &kbuf[c & 1][0];
    const __bf16* vchb = &vbuf[c & 1][0];

    for (int sub = 0; sub < KCH / 32; ++sub) {
      const int s0 = sub * 32;
      // B fragments for K^T (two 16-key tiles, contraction padded 8->32)
      v16bf bk0 = vz16(), bk1 = vz16();
      if (half == 0) {
        v8bf k0 = *(const v8bf*)(kchb + (size_t)(s0 + l16) * HD);
        v8bf k1 = *(const v8bf*)(kchb + (size_t)(s0 + 16 + l16) * HD);
#pragma unroll
        for (int j = 0; j < 8; ++j) { bk0[j] = k0[j]; bk1[j] = k1[j]; }
      }
      v8f sc0 = wmma_bf(qf, bk0, vz8());
      v8f sc1 = wmma_bf(qf, bk1, vz8());

      // online softmax; row = half*8 + r lives across 16 lanes of this half
#pragma unroll
      for (int r = 0; r < 8; ++r) {
        float mx = fmaxf(sc0[r], sc1[r]);
#pragma unroll
        for (int off = 1; off < 16; off <<= 1) mx = fmaxf(mx, __shfl_xor(mx, off, 32));
        const float mnew  = fmaxf(mrow[r], mx);
        const float scale = __expf(mrow[r] - mnew);
        const float p0 = __expf(sc0[r] - mnew);
        const float p1 = __expf(sc1[r] - mnew);
        float ps = p0 + p1;
#pragma unroll
        for (int off = 1; off < 16; off <<= 1) ps += __shfl_xor(ps, off, 32);
        lrow[r] = lrow[r] * scale + ps;
        mrow[r] = mnew;
        oacc[r] *= scale;
        const int row = half * 8 + r;
        p_lds[wave][row * 32 + l16]      = (__bf16)p0;
        p_lds[wave][row * 32 + 16 + l16] = (__bf16)p1;
      }
      asm volatile("s_wait_dscnt 0" ::: "memory");  // wave-local LDS transpose fence

      // P fragment (A 16x32) from LDS
      v16bf pa;
      {
        const int kb0 = half * 8;
        v8bf plo = *(const v8bf*)(&p_lds[wave][l16 * 32 + kb0]);
        v8bf phi = *(const v8bf*)(&p_lds[wave][l16 * 32 + kb0 + 16]);
#pragma unroll
        for (int j = 0; j < 8; ++j) { pa[j] = plo[j]; pa[j + 8] = phi[j]; }
      }
      // V fragment (B 32x16, cols 0..7 = head dims)
      v16bf vb = vz16();
      if (l16 < HD) {
        const int kb0 = half * 8;
#pragma unroll
        for (int j = 0; j < 8; ++j) {
          vb[j]     = vchb[(size_t)(s0 + kb0 + j) * HD + l16];
          vb[j + 8] = vchb[(size_t)(s0 + kb0 + 16 + j) * HD + l16];
        }
      }
      oacc = wmma_bf(pa, vb, oacc);
    }
    __syncthreads();   // all waves done with buffer c&1 before it is re-filled
  }
  // normalize + store [B,S,128] bf16 (col = h*8 + d)
  if (l16 < HD) {
    const int b = bh >> 4, h = bh & 15;
#pragma unroll
    for (int r = 0; r < 8; ++r) {
      const int row = qbase + half * 8 + r;
      attn[((size_t)(b * SEQ + row)) * DIM + h * HD + l16] = (__bf16)(oacc[r] / lrow[r]);
    }
  }
}

// ---------------- 3) output projection + residual ----------------
__global__ void __launch_bounds__(256) k_oproj(const __bf16* __restrict__ attn,
                                               const float* __restrict__ wo,
                                               const float* __restrict__ x,
                                               float* __restrict__ hout,
                                               __bf16* __restrict__ hbf) {
  const int wave = threadIdx.x >> 5, lane = threadIdx.x & 31;
  const int half = lane >> 4, l16 = lane & 15;
  const int tile = blockIdx.x * 8 + wave;
  if (tile >= (MTOT / 16) * 8) return;
  const int rt = tile & 511, ct = tile >> 9;
  const __bf16* arow = attn + (size_t)(rt * 16 + l16) * DIM;
  const int col = ct * 16 + l16;
  v8f acc = vz8();
#pragma unroll
  for (int f = 0; f < 4; ++f)
    acc = wmma_bf(afrag_bf(arow, f * 32, half),
                  bfrag_w(wo, DIM, f * 32, col, DIM, half), acc);
#pragma unroll
  for (int r = 0; r < 8; ++r) {
    const int grow = rt * 16 + half * 8 + r;
    const float hv = x[(size_t)grow * DIM + col] + acc[r];
    hout[(size_t)grow * DIM + col] = hv;
    hbf[(size_t)grow * DIM + col]  = (__bf16)hv;
  }
}

// ---------------- 4) FFN gate: silu(h@w1) * (h@w3) ----------------
__global__ void __launch_bounds__(256) k_ffn1(const __bf16* __restrict__ hbf,
                                              const float* __restrict__ w1,
                                              const float* __restrict__ w3,
                                              __bf16* __restrict__ act) {
  const int wave = threadIdx.x >> 5, lane = threadIdx.x & 31;
  const int half = lane >> 4, l16 = lane & 15;
  const int tile = blockIdx.x * 8 + wave;
  if (tile >= (MTOT / 16) * (HIDP / 16)) return;     // 512 x 22
  const int rt = tile & 511, ct = tile >> 9;
  const __bf16* hrow = hbf + (size_t)(rt * 16 + l16) * DIM;
  const int n = ct * 16 + l16;
  const int wcol = (n < HID) ? n : 0;                // clamp (pad cols zeroed on store)
  v8f a1 = vz8(), a3 = vz8();
#pragma unroll
  for (int f = 0; f < 4; ++f) {
    v16bf A = afrag_bf(hrow, f * 32, half);          // reuse A across both GEMMs
    a1 = wmma_bf(A, bfrag_w(w1, HID, f * 32, wcol, DIM, half), a1);
    a3 = wmma_bf(A, bfrag_w(w3, HID, f * 32, wcol, DIM, half), a3);
  }
#pragma unroll
  for (int r = 0; r < 8; ++r) {
    const int grow = rt * 16 + half * 8 + r;
    float v = 0.0f;
    if (n < HID) {
      const float s = a1[r];
      v = (s / (1.0f + __expf(-s))) * a3[r];         // silu * gate
    }
    act[(size_t)grow * HIDP + n] = (__bf16)v;        // pad cols -> 0
  }
}

// ---------------- 5) FFN down proj + residual -> d_out ----------------
__global__ void __launch_bounds__(256) k_ffn2(const __bf16* __restrict__ act,
                                              const float* __restrict__ w2,
                                              const float* __restrict__ hout,
                                              float* __restrict__ out) {
  const int wave = threadIdx.x >> 5, lane = threadIdx.x & 31;
  const int half = lane >> 4, l16 = lane & 15;
  const int tile = blockIdx.x * 8 + wave;
  if (tile >= (MTOT / 16) * 8) return;
  const int rt = tile & 511, ct = tile >> 9;
  const __bf16* arow = act + (size_t)(rt * 16 + l16) * HIDP;
  const int col = ct * 16 + l16;
  v8f acc = vz8();
#pragma unroll
  for (int f = 0; f < HIDP / 32; ++f)                // 11 K-steps (pad rows are 0)
    acc = wmma_bf(afrag_bf(arow, f * 32, half),
                  bfrag_w(w2, DIM, f * 32, col, HID, half), acc);
#pragma unroll
  for (int r = 0; r < 8; ++r) {
    const int grow = rt * 16 + half * 8 + r;
    out[(size_t)grow * DIM + col] = hout[(size_t)grow * DIM + col] + acc[r];
  }
}

// ---------------- launcher ----------------
extern "C" void kernel_launch(void* const* d_in, const int* in_sizes, int n_in,
                              void* d_out, int out_size, void* d_ws, size_t ws_size,
                              hipStream_t stream) {
  (void)in_sizes; (void)n_in; (void)out_size; (void)ws_size;
  const float* x  = (const float*)d_in[0];
  const float* wq = (const float*)d_in[1];
  const float* wk = (const float*)d_in[2];
  const float* wv = (const float*)d_in[3];
  const float* wo = (const float*)d_in[4];
  const float* w1 = (const float*)d_in[5];
  const float* w3 = (const float*)d_in[6];
  const float* w2 = (const float*)d_in[7];
  float* out = (float*)d_out;

  char* ws = (char*)d_ws;                       // ~19.8 MB total
  __bf16* Q    = (__bf16*)(ws);                 // 2 MB  [B,H,S,8]
  __bf16* K    = (__bf16*)(ws + (2u  << 20));   // 2 MB
  __bf16* V    = (__bf16*)(ws + (4u  << 20));   // 2 MB
  __bf16* attn = (__bf16*)(ws + (6u  << 20));   // 2 MB  [B*S,128]
  float*  hbuf = (float* )(ws + (8u  << 20));   // 4 MB  [B*S,128] f32
  __bf16* hbf  = (__bf16*)(ws + (12u << 20));   // 2 MB
  __bf16* act  = (__bf16*)(ws + (14u << 20));   // 5.8 MB [B*S,352]

  const dim3 blk(256);
  k_qkv <<<dim3((MTOT / 16) * 24 / 8), blk, 0, stream>>>(x, wq, wk, wv, Q, K, V);
  k_attn<<<dim3(SEQ / 128, BATCH * NH), blk, 0, stream>>>(Q, K, V, attn);
  k_oproj<<<dim3((MTOT / 16) * 8 / 8), blk, 0, stream>>>(attn, wo, x, hbuf, hbf);
  k_ffn1<<<dim3((MTOT / 16) * (HIDP / 16) / 8), blk, 0, stream>>>(hbf, w1, w3, act);
  k_ffn2<<<dim3((MTOT / 16) * 8 / 8), blk, 0, stream>>>(act, w2, hbuf, out);
}
